// xFormersFill_vLLMPagedAttentionGenerate_49143015801663
// MI455X (gfx1250) — matrix operand order repaired and checked by reading, not compiled, not run on hardware
//
#include <hip/hip_runtime.h>
#include <hip/hip_bf16.h>
#include <math.h>

// ---------------- problem constants (from reference) ----------------
#define H             32
#define DHEAD         128
#define ATTN_SCALE    0.08838834764831845f   // 1/sqrt(128)
#define NUM_FILL_SEQS 8
#define FILL_LEN      512
#define NUM_FILL      (NUM_FILL_SEQS * FILL_LEN)   // 4096
#define NUM_GEN       32
#define NUM_BLOCKS    1024
#define BLOCK_SIZE    16
#define MAX_BLOCKS    64
#define MAX_CTX       (MAX_BLOCKS * BLOCK_SIZE)    // 1024
#define T_TOTAL       (NUM_FILL + NUM_GEN)         // 4128
#define NUM_SLOTS     (NUM_BLOCKS * BLOCK_SIZE)    // 16384

typedef __bf16 bf16_t;
typedef __attribute__((ext_vector_type(16))) __bf16 v16bf;
typedef __attribute__((ext_vector_type(8)))  float  v8f;

static __device__ __forceinline__ bf16_t f2bf(float f) {
  return (bf16_t)f;
}

// ---------------- Tensor Data Mover (gfx1250) ----------------
#ifndef USE_TDM
#if defined(__has_builtin)
#if __has_builtin(__builtin_amdgcn_tensor_load_to_lds) && \
    __has_builtin(__builtin_amdgcn_s_wait_tensorcnt)
#define USE_TDM 1
#endif
#endif
#endif
#ifndef USE_TDM
#define USE_TDM 0
#endif

#if USE_TDM
typedef __attribute__((ext_vector_type(4))) unsigned int u32x4;
typedef __attribute__((ext_vector_type(8))) int          i32x8;
typedef __attribute__((ext_vector_type(4))) int          i32x4;

// 2-D TDM load: tile = 32 rows x 128 f32, row stride = H*DHEAD elements.
// D# packing per CDNA5 ISA sec 8.3-8.6 (groups 2/3 zero => 2-D tensor).
// This toolchain exposes the 6-arg builtin (clang-23 / therock-10.0 headers):
//   (u32x4 g0, i32x8 g1, i32x4 g2, i32x4 g3, i32x8 gx, i32 cpol)
static __device__ __forceinline__ void tdm_load_tile_f32(const float* gsrc,
                                                         unsigned lds_byte_addr)
{
  const unsigned long long ga = (unsigned long long)(uintptr_t)gsrc;
  u32x4 g0;
  g0[0] = 1u;                                              // count=1, user D#
  g0[1] = lds_byte_addr;                                   // lds_addr
  g0[2] = (unsigned)ga;                                    // global_addr[31:0]
  g0[3] = (unsigned)((ga >> 32) & 0x01FFFFFFu) | (2u << 30); // addr[56:32]|type=2

  i32x8 g1;
  g1[0] = (int)(2u << 16);                  // wg_mask=0, data_size=2 (4 bytes)
  g1[1] = (int)((unsigned)DHEAD << 16);     // tensor_dim0[15:0]  (=128)
  g1[2] = (int)((unsigned)FILL_LEN << 16);  // tensor_dim0 hi=0 | tensor_dim1 lo (=512)
  g1[3] = (int)((unsigned)DHEAD << 16);     // tensor_dim1 hi=0 | tile_dim0 (=128)
  g1[4] = 32;                               // tile_dim1=32, tile_dim2=0
  g1[5] = (int)(H * DHEAD);                 // tensor_dim0_stride lo (=4096 elems)
  g1[6] = 0;                                // stride hi / tensor_dim1_stride lo
  g1[7] = 0;
  const i32x4 gz4 = {0, 0, 0, 0};
  const i32x8 gz8 = {0, 0, 0, 0, 0, 0, 0, 0};
  __builtin_amdgcn_tensor_load_to_lds(g0, g1, gz4, gz4, gz8, 0);
}
#endif

// ---------------- inverse slot map kernels ----------------
__global__ void init_inv_kernel(int* inv) {
  int i = blockIdx.x * blockDim.x + threadIdx.x;
  if (i < NUM_SLOTS) inv[i] = -1;
}

__global__ void scatter_inv_kernel(const int* __restrict__ slot_mapping, int* inv) {
  int i = blockIdx.x * blockDim.x + threadIdx.x;
  if (i < T_TOTAL) {
    int s = slot_mapping[i];
    if (s >= 0 && s < NUM_SLOTS) inv[s] = i;
  }
}

// ---------------- prefill: causal flash attention via bf16 WMMA ----------------
// grid = (FILL_LEN/64, H, NUM_FILL_SEQS), block = 128 (4 waves)
// wave w owns query rows [blkq0 + 16w, +16); block shares K/V LDS tiles of 32 keys.
// K tiles stream in through the Tensor Data Mover, double buffered on TENSORcnt.
__launch_bounds__(128)
__global__ void prefill_attn_kernel(const float* __restrict__ q,
                                    const float* __restrict__ k,
                                    const float* __restrict__ v,
                                    float* __restrict__ out)
{
#if USE_TDM
  __shared__ float  sKf[2][32 * DHEAD];  // TDM f32 staging, double buffered 32 KB
#endif
  __shared__ bf16_t sK[32 * DHEAD];      // [tok][d]   8 KB
  __shared__ bf16_t sVt[DHEAD * 32];     // [d][tok]   8 KB (transposed V)
  __shared__ bf16_t sP[4 * 16 * 32];     // per-wave P tile, row-major [m][key] 4 KB

  const int tid  = threadIdx.x;
  const int wave = tid >> 5;
  const int lane = tid & 31;
  const int hf   = lane >> 4;           // which 16-lane half
  const int l16  = lane & 15;

  const int seq   = blockIdx.z;
  const int h     = blockIdx.y;
  const int blkq0 = blockIdx.x * 64;
  const int q0    = blkq0 + wave * 16;  // wave's first query row (within seq)
  const int tq    = seq * FILL_LEN;     // token offset of this sequence

  // ---- Q A-fragments (16x128 -> 4 chunks of 16x32), f32 -> bf16 ----
  // A layout: lane holds row M=l16; K = c*32 + 16*hf + [0..16)
  v16bf qa[4];
  {
    const float* qp = q + ((size_t)(tq + q0 + l16) * H + h) * DHEAD;
    #pragma unroll
    for (int c = 0; c < 4; ++c) {
      const int d0 = 32 * c + 16 * hf;
      #pragma unroll
      for (int e = 0; e < 8; ++e) {
        qa[c][e]     = f2bf(qp[d0 + e]);
        qa[c][8 + e] = f2bf(qp[d0 + 8 + e]);
      }
    }
  }

  // ---- accumulators ----
  v8f O[8];
  #pragma unroll
  for (int j = 0; j < 8; ++j)
    #pragma unroll
    for (int r = 0; r < 8; ++r) O[j][r] = 0.0f;

  float mrow[8], lrow[8];
  #pragma unroll
  for (int r = 0; r < 8; ++r) { mrow[r] = -1.0e30f; lrow[r] = 0.0f; }

  const int nkb = blkq0 / 32 + 2;       // 32-key blocks covering keys <= blkq0+63

#if USE_TDM
  const float* kseq = k + ((size_t)tq * H + h) * DHEAD;   // seq token 0, head h
  const unsigned ldsK0 = (unsigned)(uintptr_t)(&sKf[0][0]);
  const unsigned ldsK1 = (unsigned)(uintptr_t)(&sKf[1][0]);
  if (wave == 0) {
    tdm_load_tile_f32(kseq, ldsK0);     // prologue: K block 0 in flight
  }
#endif

  for (int kb = 0; kb < nkb; ++kb) {
    const int kstart = kb * 32;

    __syncthreads();  // previous iteration done reading sK/sVt

#if USE_TDM
    if (wave == 0) {
      if (kb + 1 < nkb) {
        // stream next K tile while we consume the current one
        tdm_load_tile_f32(kseq + (size_t)(kb + 1) * 32 * H * DHEAD,
                          ((kb + 1) & 1) ? ldsK1 : ldsK0);
        __builtin_amdgcn_s_wait_tensorcnt((short)1);   // current tile landed
      } else {
        __builtin_amdgcn_s_wait_tensorcnt((short)0);
      }
    }
    __syncthreads();  // publish TDM-written stage to all waves

    // ---- convert K stage (f32 LDS) -> bf16 sK; load V global -> bf16 sVt ----
    {
      const int tok   = tid >> 2;            // 0..31
      const int dbase = (tid & 3) * 32;      // 0,32,64,96
      const float* kst = ((kb & 1) ? &sKf[1][0] : &sKf[0][0]) + tok * DHEAD + dbase;
      const float* vp  = v + ((size_t)(tq + kstart + tok) * H + h) * DHEAD + dbase;
      #pragma unroll
      for (int i = 0; i < 32; i += 4) {
        const float4 kk = *(const float4*)(kst + i);
        const float4 vv = *(const float4*)(vp + i);
        const int d = dbase + i;
        sK[tok * DHEAD + d + 0] = f2bf(kk.x);
        sK[tok * DHEAD + d + 1] = f2bf(kk.y);
        sK[tok * DHEAD + d + 2] = f2bf(kk.z);
        sK[tok * DHEAD + d + 3] = f2bf(kk.w);
        sVt[(d + 0) * 32 + tok] = f2bf(vv.x);
        sVt[(d + 1) * 32 + tok] = f2bf(vv.y);
        sVt[(d + 2) * 32 + tok] = f2bf(vv.z);
        sVt[(d + 3) * 32 + tok] = f2bf(vv.w);
      }
      if (kb + 1 < nkb) {
        __builtin_prefetch(vp + (size_t)32 * H * DHEAD, 0, 1);  // next V block
      }
    }
#else
    // ---- fallback: cooperative K/V tile load from global, f32 -> bf16 ----
    {
      const int tok   = tid >> 2;
      const int dbase = (tid & 3) * 32;
      const size_t rowoff = ((size_t)(tq + kstart + tok) * H + h) * DHEAD + dbase;
      const float* kp = k + rowoff;
      const float* vp = v + rowoff;
      #pragma unroll
      for (int i = 0; i < 32; i += 4) {
        const float4 kk = *(const float4*)(kp + i);
        const float4 vv = *(const float4*)(vp + i);
        const int d = dbase + i;
        sK[tok * DHEAD + d + 0] = f2bf(kk.x);
        sK[tok * DHEAD + d + 1] = f2bf(kk.y);
        sK[tok * DHEAD + d + 2] = f2bf(kk.z);
        sK[tok * DHEAD + d + 3] = f2bf(kk.w);
        sVt[(d + 0) * 32 + tok] = f2bf(vv.x);
        sVt[(d + 1) * 32 + tok] = f2bf(vv.y);
        sVt[(d + 2) * 32 + tok] = f2bf(vv.z);
        sVt[(d + 3) * 32 + tok] = f2bf(vv.w);
      }
      if (kb + 1 < nkb) {
        __builtin_prefetch(kp + (size_t)32 * H * DHEAD, 0, 1);
        __builtin_prefetch(vp + (size_t)32 * H * DHEAD, 0, 1);
      }
    }
#endif
    __syncthreads();

    if (kstart <= q0 + 15) {              // wave-uniform: EXEC stays all-1s
      // ---- S = Q * K^T : two 16x16 N-tiles, K-dim 128 via 4 chained WMMAs ----
      v8f st0 = {0.f,0.f,0.f,0.f,0.f,0.f,0.f,0.f};
      v8f st1 = {0.f,0.f,0.f,0.f,0.f,0.f,0.f,0.f};
      #pragma unroll
      for (int c = 0; c < 4; ++c) {
        // B layout: lane holds col N=l16 (key), K = c*32 + 16*hf + [0..16)
        v16bf kb0, kb1;
        const bf16_t* p0 = &sK[(l16)      * DHEAD + 32 * c + 16 * hf];
        const bf16_t* p1 = &sK[(16 + l16) * DHEAD + 32 * c + 16 * hf];
        #pragma unroll
        for (int e = 0; e < 16; ++e) { kb0[e] = p0[e]; kb1[e] = p1[e]; }
        st0 = __builtin_amdgcn_wmma_f32_16x16x32_bf16(false, qa[c], false, kb0,
                                                      (short)0, st0, false, false);
        st1 = __builtin_amdgcn_wmma_f32_16x16x32_bf16(false, qa[c], false, kb1,
                                                      (short)0, st1, false, false);
      }

      // ---- online softmax (rows live on 16-lane groups; reduce via shfl_xor) ----
      float alpha[8];
      #pragma unroll
      for (int r = 0; r < 8; ++r) {
        const int qrow = q0 + r + 8 * hf;
        float s0 = st0[r] * ATTN_SCALE;
        float s1 = st1[r] * ATTN_SCALE;
        if (kstart + l16 > qrow)      s0 = -3.0e38f;   // causal mask
        if (kstart + 16 + l16 > qrow) s1 = -3.0e38f;
        float mx = fmaxf(s0, s1);
        #pragma unroll
        for (int off = 8; off >= 1; off >>= 1) mx = fmaxf(mx, __shfl_xor(mx, off));
        const float mnew = fmaxf(mrow[r], mx);
        const float a    = __expf(mrow[r] - mnew);
        mrow[r] = mnew; alpha[r] = a;
        const float p0 = (s0 < -1.0e29f) ? 0.0f : __expf(s0 - mnew);
        const float p1 = (s1 < -1.0e29f) ? 0.0f : __expf(s1 - mnew);
        st0[r] = p0; st1[r] = p1;
        float ps = p0 + p1;
        #pragma unroll
        for (int off = 8; off >= 1; off >>= 1) ps += __shfl_xor(ps, off);
        lrow[r] = lrow[r] * a + ps;
      }
      #pragma unroll
      for (int j = 0; j < 8; ++j)
        #pragma unroll
        for (int r = 0; r < 8; ++r) O[j][r] *= alpha[r];

      // ---- re-layout P (C-layout -> A-layout) through per-wave LDS tile ----
      bf16_t* pw = &sP[wave * 16 * 32];
      #pragma unroll
      for (int r = 0; r < 8; ++r) {
        const int row = r + 8 * hf;
        pw[row * 32 + l16]      = f2bf(st0[r]);
        pw[row * 32 + 16 + l16] = f2bf(st1[r]);
      }
      // same-wave LDS ops are in-order: store -> load is safe without barrier
      v16bf pa;
      {
        const bf16_t* pr = &pw[l16 * 32];
        const int kbase = 16 * hf;
        #pragma unroll
        for (int e = 0; e < 8; ++e) {
          pa[e]     = pr[kbase + e];
          pa[8 + e] = pr[kbase + 8 + e];
        }
      }

      // ---- O += P * V : 8 d-tiles, K-dim 32 ----
      #pragma unroll
      for (int j = 0; j < 8; ++j) {
        v16bf vb;   // B layout: col N = d = 16j+l16, K = tok = 16*hf + [0..16)
        const bf16_t* vp2 = &sVt[(16 * j + l16) * 32 + 16 * hf];
        #pragma unroll
        for (int e = 0; e < 16; ++e) vb[e] = vp2[e];
        O[j] = __builtin_amdgcn_wmma_f32_16x16x32_bf16(false, pa, false, vb,
                                                       (short)0, O[j], false, false);
      }
    }
  }

  // ---- epilogue: normalize and store f32 output ----
  float linv[8];
  #pragma unroll
  for (int r = 0; r < 8; ++r) linv[r] = 1.0f / lrow[r];

  #pragma unroll
  for (int j = 0; j < 8; ++j) {
    #pragma unroll
    for (int r = 0; r < 8; ++r) {
      const int row = tq + q0 + r + 8 * hf;           // token index
      out[(size_t)row * (H * DHEAD) + h * DHEAD + 16 * j + l16] = O[j][r] * linv[r];
    }
  }
}

// ---------------- decode: paged attention (memory-bound gather) ----------------
// grid = (NUM_GEN, H), block = 256 (8 waves).
// Resolved V row pointers are cached in LDS once; P*V is split over
// 2 key-groups x 128 dims to double the number of loads in flight.
__launch_bounds__(256)
__global__ void gen_attn_kernel(const float* __restrict__ q,
                                const float* __restrict__ k,
                                const float* __restrict__ v,
                                const float* __restrict__ k_cache,
                                const float* __restrict__ v_cache,
                                const int*   __restrict__ block_tables,
                                const int*   __restrict__ context_lens,
                                const int*   __restrict__ inv,
                                float* __restrict__ out)
{
  __shared__ float        sS[MAX_CTX];     // scores / probabilities   4 KB
  __shared__ const float* sVP[MAX_CTX];    // resolved V row pointers  8 KB
  __shared__ float        sQ[DHEAD];
  __shared__ float        red[8];
  __shared__ float        part[2][DHEAD];  // P*V partial sums         1 KB

  const int g   = blockIdx.x;
  const int h   = blockIdx.y;
  const int tid = threadIdx.x;

  int ctx = context_lens[g];
  if (ctx < 1) ctx = 1;
  if (ctx > MAX_CTX) ctx = MAX_CTX;

  const float* qp = q + ((size_t)(NUM_FILL + g) * H + h) * DHEAD;
  if (tid < DHEAD) sQ[tid] = qp[tid];
  __syncthreads();

  // ---- scores + pointer resolution (block_tables/inv read once per key) ----
  for (int j = tid; j < ctx; j += blockDim.x) {
    const int slot = block_tables[g * MAX_BLOCKS + (j >> 4)] * BLOCK_SIZE + (j & 15);
    const int src  = inv[slot];
    const float* kp = (src >= 0) ? (k       + ((size_t)src  * H + h) * DHEAD)
                                 : (k_cache + ((size_t)slot * H + h) * DHEAD);
    sVP[j] = (src >= 0) ? (v       + ((size_t)src  * H + h) * DHEAD)
                        : (v_cache + ((size_t)slot * H + h) * DHEAD);
    float acc = 0.0f;
    #pragma unroll 8
    for (int d = 0; d < DHEAD; d += 4) {
      const float4 kk = *(const float4*)(kp + d);
      acc += kk.x * sQ[d] + kk.y * sQ[d + 1] + kk.z * sQ[d + 2] + kk.w * sQ[d + 3];
    }
    sS[j] = acc * ATTN_SCALE;
  }
  __syncthreads();

  // ---- block max ----
  float mx = -3.0e38f;
  for (int j = tid; j < ctx; j += blockDim.x) mx = fmaxf(mx, sS[j]);
  #pragma unroll
  for (int off = 16; off >= 1; off >>= 1) mx = fmaxf(mx, __shfl_xor(mx, off));
  if ((tid & 31) == 0) red[tid >> 5] = mx;
  __syncthreads();
  #pragma unroll
  for (int w = 0; w < 8; ++w) mx = fmaxf(mx, red[w]);
  __syncthreads();

  // ---- exp + sum ----
  float sum = 0.0f;
  for (int j = tid; j < ctx; j += blockDim.x) {
    const float p = __expf(sS[j] - mx);
    sS[j] = p;
    sum += p;
  }
  #pragma unroll
  for (int off = 16; off >= 1; off >>= 1) sum += __shfl_xor(sum, off);
  __syncthreads();
  if ((tid & 31) == 0) red[tid >> 5] = sum;
  __syncthreads();
  sum = 0.0f;
  #pragma unroll
  for (int w = 0; w < 8; ++w) sum += red[w];
  const float isum = 1.0f / sum;

  // ---- weighted V gather: (key-group, dim) = (tid>>7, tid&127) ----
  {
    const int grp = tid >> 7;          // 0..1
    const int d   = tid & (DHEAD - 1); // 0..127  (coalesced across lanes)
    float acc = 0.0f;
    for (int j = grp; j < ctx; j += 2) {
      acc += sS[j] * sVP[j][d];
    }
    part[grp][d] = acc;
  }
  __syncthreads();
  if (tid < DHEAD) {
    out[(size_t)(NUM_FILL + g) * (H * DHEAD) + h * DHEAD + tid] =
        (part[0][tid] + part[1][tid]) * isum;
  }
}

// ---------------- launcher ----------------
extern "C" void kernel_launch(void* const* d_in, const int* in_sizes, int n_in,
                              void* d_out, int out_size, void* d_ws, size_t ws_size,
                              hipStream_t stream)
{
  const float* q            = (const float*)d_in[0];
  const float* k            = (const float*)d_in[1];
  const float* v            = (const float*)d_in[2];
  const float* k_cache      = (const float*)d_in[3];
  const float* v_cache      = (const float*)d_in[4];
  const int*   slot_mapping = (const int*)d_in[5];
  const int*   block_tables = (const int*)d_in[6];
  const int*   context_lens = (const int*)d_in[7];
  float* out = (float*)d_out;

  int* inv = (int*)d_ws;   // 16384 ints = 64 KB inverse slot map

  init_inv_kernel<<<(NUM_SLOTS + 255) / 256, 256, 0, stream>>>(inv);
  scatter_inv_kernel<<<(T_TOTAL + 255) / 256, 256, 0, stream>>>(slot_mapping, inv);

  dim3 gp(FILL_LEN / 64, H, NUM_FILL_SEQS);   // (8, 32, 8)
  prefill_attn_kernel<<<gp, 128, 0, stream>>>(q, k, v, out);

  dim3 gg(NUM_GEN, H);                        // (32, 32)
  gen_attn_kernel<<<gg, 256, 0, stream>>>(q, k, v, k_cache, v_cache,
                                          block_tables, context_lens, inv, out);
}
